// GraphSparseMoe_55525337203005
// MI455X (gfx1250) — compile-verified, matrix operand order checked by reading, not run on hardware
//
#include <hip/hip_runtime.h>
#include <hip/hip_bf16.h>
#include <math.h>
#include <stdint.h>

// ---------------------------------------------------------------------------
// Problem dims (fixed by the reference)
// ---------------------------------------------------------------------------
#define T_TOK 2048      // B*S tokens
#define DIM   1024      // D
#define NEXP  8         // E
#define DGATE 256       // DG
#define HID   2048      // H
#define NPAD  16        // padded node count (E + token node = 9 -> 16)

// GEMM workgroup tiling: 256 threads = 8 waves, 128(M) x 64(N) block, K-step 32
#define BM 128
#define BN 64
#define BK 32

typedef __attribute__((ext_vector_type(16))) __bf16 v16bf;
typedef __attribute__((ext_vector_type(8)))  __bf16 v8bf;
typedef __attribute__((ext_vector_type(8)))  float  v8f;

__device__ __forceinline__ float silu_f(float x) { return x / (1.f + __expf(-x)); }

// s_wait_asynccnt 0 (async-to-LDS fence)
__device__ __forceinline__ void wait_async0() {
#if __has_builtin(__builtin_amdgcn_s_wait_asynccnt)
    __builtin_amdgcn_s_wait_asynccnt(0);
#else
    asm volatile("s_wait_asynccnt 0" ::: "memory");
#endif
}

// One 16-byte global->LDS async copy per lane (ASYNCcnt-tracked, no VGPR data).
__device__ __forceinline__ void async_copy_b128(const void* gptr, void* lptr) {
    unsigned lds_off = (unsigned)(uintptr_t)lptr;           // addr[31:0] == LDS offset
    unsigned long long ga = (unsigned long long)(uintptr_t)gptr;
    asm volatile("global_load_async_to_lds_b128 %0, %1, off"
                 :: "v"(lds_off), "v"(ga)
                 : "memory");
}

// ---------------------------------------------------------------------------
// WMMA fragment layout (wave32, 16x16x32 bf16):
//   lane L: m/n = L&15 ; kb = (L<16)?0:8
//   elem j<8  -> K = kb+j        (contiguous 8 -> one 16B LDS load)
//   elem j>=8 -> K = 16+kb+(j-8) (contiguous 8 -> one 16B LDS load)
// C/D (v8f): lane L: n = L&15 ; VGPR r -> M = (L<16) ? r : 8+r
// ---------------------------------------------------------------------------
__device__ __forceinline__ v16bf frag_from_lds(const __bf16* row_base, int kb) {
    v8bf lo = *(const v8bf*)(row_base + kb);
    v8bf hi = *(const v8bf*)(row_base + 16 + kb);
    return __builtin_shufflevector(lo, hi, 0,1,2,3,4,5,6,7,8,9,10,11,12,13,14,15);
}

// Direct-from-global A loader (GCN kernel: contiguous bf16 rows -> b128 loads)
__device__ __forceinline__ v16bf load_a_bf16(const __bf16* __restrict__ p, int lda, int lane) {
    int m  = lane & 15;
    int kb = (lane < 16) ? 0 : 8;
    const __bf16* r = p + (size_t)m * lda;
    v8bf lo = *(const v8bf*)(r + kb);
    v8bf hi = *(const v8bf*)(r + 16 + kb);
    return __builtin_shufflevector(lo, hi, 0,1,2,3,4,5,6,7,8,9,10,11,12,13,14,15);
}

// ---------------------------------------------------------------------------
// K0: fp32 -> bf16 convert of tokens + zero the `final` output region
// ---------------------------------------------------------------------------
__global__ void init_kernel(const float* __restrict__ x, __bf16* __restrict__ x_bf,
                            float* __restrict__ final_out) {
    size_t idx = (size_t)blockIdx.x * blockDim.x + threadIdx.x;
    if (idx < (size_t)T_TOK * DIM) {
        x_bf[idx]      = (__bf16)x[idx];
        final_out[idx] = 0.f;
    }
}

// ---------------------------------------------------------------------------
// K2: expg = silu(X @ W_struct)  [E=8, DG=256]  (tiny)
// ---------------------------------------------------------------------------
__global__ void expg_kernel(const float* __restrict__ X, const float* __restrict__ Wst,
                            __bf16* __restrict__ expg) {
    int g = threadIdx.x;   // 0..255
    int e = blockIdx.x;    // 0..7
    float s = 0.f;
    for (int d = 0; d < DIM; ++d)
        s += X[(size_t)e * DIM + d] * Wst[(size_t)d * DGATE + g];
    expg[(size_t)e * DGATE + g] = (__bf16)silu_f(s);
}

// ---------------------------------------------------------------------------
// Tiled WMMA GEMM with double-buffered LDS pipeline.
//   C(MxN) = A(MxK bf16) * B(KxN fp32 -> bf16)
// Per 256-thread workgroup: 128x64 output block, K-step 32.
//  - A tile (128x32 bf16, 8KB): global_load_async_to_lds_b128 (ASYNCcnt)
//  - B panel (32x64 f32): 8 pipelined b32 loads/thread -> 4 packed cvts ->
//    one ds_store_b128 into N-major LDS
//  - stage(next) overlaps with WMMA compute on the current buffer
//  - all 4 B fragments loaded before the 4 back-to-back WMMAs (one DS fence)
// mode 0: out_bf = bf16(silu(acc))
// mode 1: out_f += wfull[row*8+expert] * acc
// ---------------------------------------------------------------------------
__global__ __launch_bounds__(256)
void wmma_gemm_tiled(const __bf16* __restrict__ A, int lda,
                     const float*  __restrict__ B, int ldb,
                     int N, int K, int mode,
                     __bf16* __restrict__ out_bf,
                     float*  __restrict__ out_f,
                     const float* __restrict__ wfull, int expert, int ldo) {
    __shared__ __attribute__((aligned(16))) __bf16 shA[2][BM * BK];   // row-major [128][32]
    __shared__ __attribute__((aligned(16))) __bf16 shB[2][BN * BK];   // N-major   [64][32]

    int tid  = threadIdx.x;
    int lane = tid & 31;
    int wv   = tid >> 5;                 // 0..7
    int nbw  = N / BN;
    int wgM  = blockIdx.x / nbw;
    int wgN  = blockIdx.x % nbw;
    int n0   = wgN * BN;

    int m  = lane & 15;
    int kb = (lane < 16) ? 0 : 8;

    // staging coordinates
    int bn = tid & 63;                   // B: column within panel
    int bk = (tid >> 6) * 8;             // B: k-strip base (0,8,16,24)

    auto stage = [&](int buf, int k0) {
        // --- A tile: 512 x 16B chunks, async global->LDS ---
#pragma unroll
        for (int i = 0; i < 2; ++i) {
            int c   = tid + 256 * i;
            int row = c >> 2;
            int seg = c & 3;
            async_copy_b128(A + (size_t)(wgM * BM + row) * lda + k0 + seg * 8,
                            &shA[buf][row * BK + seg * 8]);
        }
        // --- B panel: 8 independent b32 loads, pack, one ds_store_b128 ---
        const float* colp = B + (size_t)(k0 + bk) * ldb + n0 + bn;
        float f[8];
#pragma unroll
        for (int i = 0; i < 8; ++i) f[i] = colp[(size_t)i * ldb];
        v8bf pk;
#pragma unroll
        for (int i = 0; i < 8; ++i) pk[i] = (__bf16)f[i];
        *(v8bf*)&shB[buf][bn * BK + bk] = pk;
    };

    v8f acc[4];
#pragma unroll
    for (int s = 0; s < 4; ++s) acc[s] = (v8f){0.f,0.f,0.f,0.f,0.f,0.f,0.f,0.f};

    int nsteps = K / BK;
    stage(0, 0);
    for (int s = 0; s < nsteps; ++s) {
        int buf = s & 1;
        wait_async0();                   // my async chunks for `buf` landed
        __syncthreads();                 // everyone's A+B for `buf` visible;
                                         // everyone done reading buf^1
        if (s + 1 < nsteps)
            stage(buf ^ 1, (s + 1) * BK);   // prefetch next while computing

        // load A + all 4 B fragments first, then 4 independent WMMAs
        v16bf a = frag_from_lds(&shA[buf][(wv * 16 + m) * BK], kb);
        v16bf bfr[4];
#pragma unroll
        for (int t4 = 0; t4 < 4; ++t4)
            bfr[t4] = frag_from_lds(&shB[buf][(t4 * 16 + (lane & 15)) * BK], kb);
#pragma unroll
        for (int t4 = 0; t4 < 4; ++t4)
            acc[t4] = __builtin_amdgcn_wmma_f32_16x16x32_bf16(false, a, false, bfr[t4],
                                                              (short)0, acc[t4], false, false);
    }

    // --- epilogue ---
    int mw0 = wgM * BM + wv * 16;
#pragma unroll
    for (int s = 0; s < 4; ++s) {
        int n = n0 + s * 16 + (lane & 15);
#pragma unroll
        for (int r = 0; r < 8; ++r) {
            int row = mw0 + ((lane < 16) ? r : 8 + r);
            float v = acc[s][r];
            if (mode == 0) {
                out_bf[(size_t)row * ldo + n] = (__bf16)silu_f(v);
            } else {
                float w = wfull[(size_t)row * NEXP + expert];
                out_f[(size_t)row * ldo + n] += w * v;
            }
        }
    }
}

// ---------------------------------------------------------------------------
// K3: node_x init: node[t, n, g] = n<8 ? expg[n,g] : (n==8 ? xg[t,g] : 0)
// ---------------------------------------------------------------------------
__global__ void node_init_kernel(const __bf16* __restrict__ expg,
                                 const __bf16* __restrict__ xg,
                                 __bf16* __restrict__ node) {
    size_t idx = (size_t)blockIdx.x * blockDim.x + threadIdx.x;
    if (idx >= (size_t)T_TOK * NPAD * DGATE) return;
    int    g = (int)(idx & (DGATE - 1));
    int    n = (int)((idx >> 8) & (NPAD - 1));
    size_t t = idx >> 12;
    __bf16 v = (__bf16)0.f;
    if (n < NEXP)       v = expg[(size_t)n * DGATE + g];
    else if (n == NEXP) v = xg[t * DGATE + g];
    node[idx] = v;
}

// ---------------------------------------------------------------------------
// K4: one GCN layer. 128 threads = 4 waves = 4 tokens per block.
//   Z           = node_in[t] (16x256) @ conv_W[l] (256x256)      [WMMA 1]
//   node_out[t] = bf16( silu( A_pad(16x16) @ Z ) )               [WMMA 2]
// - token's 8 A-fragments preloaded to registers, reused over 16 N-tiles
// - conv_W N-panel (256x16) staged once per block into N-major bf16 LDS
// - two interleaved accumulators halve the WMMA RAW chain
// - WMMA-1 C-layout maps lane-for-lane onto WMMA-2 B-layout
// ---------------------------------------------------------------------------
__global__ __launch_bounds__(128)
void gcn_layer_kernel(const __bf16* __restrict__ node_in,
                      const float*  __restrict__ convW,   // 256x256
                      const float*  __restrict__ Ahat,    // 9x9
                      __bf16* __restrict__ node_out) {
    __shared__ float Ah[81];
    __shared__ __attribute__((aligned(16))) __bf16 shBn[16 * DGATE];   // N-major [16][256], 8KB

    int tid = threadIdx.x;
    if (tid < 81) Ah[tid] = Ahat[tid];

    int lane = tid & 31;
    int wv   = tid >> 5;                 // 0..3
    int t    = blockIdx.x * 4 + wv;      // token
    int m    = lane & 15;
    int kb   = (lane < 16) ? 0 : 8;

    // Preload this token's 8 A fragments (16x32 bf16 tiles, contiguous rows)
    const __bf16* An = node_in + (size_t)t * NPAD * DGATE;
    v16bf afr[8];
#pragma unroll
    for (int ks = 0; ks < 8; ++ks) afr[ks] = load_a_bf16(An + ks * 32, DGATE, lane);

    __syncthreads();                     // Ah ready

    // A_pad fragment: A_pad[m][k] = (m<9 && k<9) ? A_hat[m*9+k] : 0
    v16bf apad;
#pragma unroll
    for (int j = 0; j < 8; ++j) {
        int k = kb + j;
        apad[j] = (m < 9 && k < 9) ? (__bf16)Ah[m * 9 + k] : (__bf16)0.f;
    }
#pragma unroll
    for (int j = 0; j < 8; ++j) apad[8 + j] = (__bf16)0.f;

    // B-panel staging coords: thread -> (n-local, k-strip)
    int nl = tid >> 3;                   // 0..15
    int kh = (tid & 7) * 32;             // 0..224

    for (int nt = 0; nt < (DGATE / 16); ++nt) {
        __syncthreads();                 // previous panel's readers done
        // stage conv_W[:, nt*16 .. +16) as N-major bf16
        {
            const float* bp = convW + nt * 16 + nl;
#pragma unroll
            for (int q = 0; q < 4; ++q) {
                float f[8];
#pragma unroll
                for (int i = 0; i < 8; ++i)
                    f[i] = bp[(size_t)(kh + q * 8 + i) * DGATE];
                v8bf pk;
#pragma unroll
                for (int i = 0; i < 8; ++i) pk[i] = (__bf16)f[i];
                *(v8bf*)&shBn[nl * DGATE + kh + q * 8] = pk;
            }
        }
        __syncthreads();

        v8f acc0 = {0.f,0.f,0.f,0.f,0.f,0.f,0.f,0.f};
        v8f acc1 = {0.f,0.f,0.f,0.f,0.f,0.f,0.f,0.f};
#pragma unroll
        for (int ks = 0; ks < 8; ks += 2) {
            v16bf b0 = frag_from_lds(&shBn[(lane & 15) * DGATE + ks * 32], kb);
            v16bf b1 = frag_from_lds(&shBn[(lane & 15) * DGATE + (ks + 1) * 32], kb);
            acc0 = __builtin_amdgcn_wmma_f32_16x16x32_bf16(false, afr[ks],     false, b0,
                                                           (short)0, acc0, false, false);
            acc1 = __builtin_amdgcn_wmma_f32_16x16x32_bf16(false, afr[ks + 1], false, b1,
                                                           (short)0, acc1, false, false);
        }

        // graph mix: accumulator re-fed as B of the A_pad WMMA (same-lane map)
        v16bf b2;
#pragma unroll
        for (int j = 0; j < 8; ++j) b2[j]     = (__bf16)(acc0[j] + acc1[j]);
#pragma unroll
        for (int j = 0; j < 8; ++j) b2[8 + j] = (__bf16)0.f;

        v8f z = {0.f,0.f,0.f,0.f,0.f,0.f,0.f,0.f};
        z = __builtin_amdgcn_wmma_f32_16x16x32_bf16(false, apad, false, b2,
                                                    (short)0, z, false, false);
        int col = nt * 16 + (lane & 15);
#pragma unroll
        for (int r = 0; r < 8; ++r) {
            int node = (lane < 16) ? r : 8 + r;
            node_out[(size_t)t * NPAD * DGATE + (size_t)node * DGATE + col] =
                (__bf16)silu_f(z[r]);
        }
    }
}

// ---------------------------------------------------------------------------
// K5: router — logits, softmax over 8 experts, top-2, routing weights.
// ---------------------------------------------------------------------------
__global__ void router_kernel(const __bf16* __restrict__ nodeF,
                              const float*  __restrict__ w_proj,
                              float* __restrict__ probs_out,   // [T, E]
                              float* __restrict__ w_full,      // [T, E]
                              float* __restrict__ rw_un,       // [T, 2]
                              int*   __restrict__ sel_out) {   // [T, 2]
    int t = blockIdx.x * blockDim.x + threadIdx.x;
    if (t >= T_TOK) return;
    const __bf16* nx = nodeF + (size_t)t * NPAD * DGATE;

    float logits[NEXP];
#pragma unroll
    for (int e = 0; e < NEXP; ++e) {
        float s = 0.f;
        for (int g = 0; g < DGATE; ++g)
            s += (float)nx[(size_t)e * DGATE + g] * w_proj[g];
        logits[e] = s;
    }
    float mx = logits[0];
#pragma unroll
    for (int e = 1; e < NEXP; ++e) mx = fmaxf(mx, logits[e]);
    float p[NEXP], sum = 0.f;
#pragma unroll
    for (int e = 0; e < NEXP; ++e) { p[e] = __expf(logits[e] - mx); sum += p[e]; }
    float inv = 1.f / sum;
#pragma unroll
    for (int e = 0; e < NEXP; ++e) p[e] *= inv;

    int i1 = 0;
#pragma unroll
    for (int e = 1; e < NEXP; ++e) if (p[e] > p[i1]) i1 = e;
    int i2 = (i1 == 0) ? 1 : 0;
#pragma unroll
    for (int e = 0; e < NEXP; ++e) if (e != i1 && p[e] > p[i2]) i2 = e;

    float r1 = p[i1], r2 = p[i2], s2 = r1 + r2;
#pragma unroll
    for (int e = 0; e < NEXP; ++e) {
        probs_out[(size_t)t * NEXP + e] = p[e];
        w_full[(size_t)t * NEXP + e] =
            (e == i1) ? (r1 / s2) : ((e == i2) ? (r2 / s2) : 0.f);
    }
    rw_un[2 * t] = r1;   rw_un[2 * t + 1] = r2;
    sel_out[2 * t] = i1; sel_out[2 * t + 1] = i2;
}

// ---------------------------------------------------------------------------
// K8: deterministic count accumulation + lamb/theta rows of loss_component
// ---------------------------------------------------------------------------
__global__ void finalize_kernel(const float* __restrict__ count,
                                const float* __restrict__ lamb,
                                const float* __restrict__ theta,
                                const int*   __restrict__ topk,
                                const float* __restrict__ rw_un,
                                const int*   __restrict__ sel,
                                float* __restrict__ loss_out) {
    int e = threadIdx.x;
    if (e >= NEXP) return;
    float tk  = (float)(*topk);
    float acc = count[e];
    for (int t = 0; t < T_TOK; ++t) {
        if (sel[2 * t]     == e) acc += rw_un[2 * t]     * tk;
        if (sel[2 * t + 1] == e) acc += rw_un[2 * t + 1] * tk;
    }
    loss_out[(size_t)T_TOK * NEXP + e]       = acc;
    loss_out[(size_t)(T_TOK + 1) * NEXP + e] = lamb[0];
    loss_out[(size_t)(T_TOK + 2) * NEXP + e] = theta[0];
}

// ---------------------------------------------------------------------------
// Host launcher
// ---------------------------------------------------------------------------
extern "C" void kernel_launch(void* const* d_in, const int* in_sizes, int n_in,
                              void* d_out, int out_size, void* d_ws, size_t ws_size,
                              hipStream_t stream) {
    const float* x        = (const float*)d_in[0];
    const float* X        = (const float*)d_in[1];
    const float* W_mlp    = (const float*)d_in[2];
    const float* W_struct = (const float*)d_in[3];
    const float* conv_W   = (const float*)d_in[4];
    const float* w_proj   = (const float*)d_in[5];
    const float* W1       = (const float*)d_in[6];
    const float* W2       = (const float*)d_in[7];
    const float* lamb     = (const float*)d_in[8];
    const float* theta    = (const float*)d_in[9];
    const float* count    = (const float*)d_in[10];
    const float* A_hat    = (const float*)d_in[11];
    const int*   topk     = (const int*)d_in[12];

    float* final_out = (float*)d_out;
    float* loss_out  = (float*)d_out + (size_t)T_TOK * DIM;

    char* ws = (char*)d_ws;
    size_t off = 0;
    __bf16* x_bf  = (__bf16*)(ws + off); off += (size_t)T_TOK * DIM * 2;
    __bf16* xg_bf = (__bf16*)(ws + off); off += (size_t)T_TOK * DGATE * 2;
    __bf16* expg  = (__bf16*)(ws + off); off += (size_t)NEXP * DGATE * 2 + 256;
    __bf16* nodeA = (__bf16*)(ws + off); off += (size_t)T_TOK * NPAD * DGATE * 2;
    __bf16* nodeB = (__bf16*)(ws + off); off += (size_t)T_TOK * NPAD * DGATE * 2;
    __bf16* h_bf  = (__bf16*)(ws + off); off += (size_t)T_TOK * HID * 2;
    float*  wfull = (float*)(ws + off);  off += (size_t)T_TOK * NEXP * 4;
    float*  rw_un = (float*)(ws + off);  off += (size_t)T_TOK * 2 * 4;
    int*    sel   = (int*)(ws + off);    off += (size_t)T_TOK * 2 * 4;
    (void)ws_size; (void)n_in; (void)in_sizes; (void)out_size;

    // K0: bf16 convert tokens + zero final output
    init_kernel<<<(T_TOK * DIM + 255) / 256, 256, 0, stream>>>(x, x_bf, final_out);

    // K2: expert structural embeddings
    expg_kernel<<<NEXP, DGATE, 0, stream>>>(X, W_struct, expg);

    // K1: gate MLP xg = silu(x @ W_mlp)   [2048 x 256 x 1024]
    wmma_gemm_tiled<<<(T_TOK / BM) * (DGATE / BN), 256, 0, stream>>>(
        x_bf, DIM, W_mlp, DGATE, DGATE, DIM,
        /*mode=*/0, xg_bf, nullptr, nullptr, 0, DGATE);

    // K3: assemble node tensor [T, 16, 256]
    node_init_kernel<<<(int)(((size_t)T_TOK * NPAD * DGATE + 255) / 256), 256, 0, stream>>>(
        expg, xg_bf, nodeA);

    // K4: three GCN layers (A -> B -> A -> B), 4 tokens per block
    {
        int blocks = T_TOK / 4;
        gcn_layer_kernel<<<blocks, 128, 0, stream>>>(nodeA, conv_W + 0 * DGATE * DGATE, A_hat, nodeB);
        gcn_layer_kernel<<<blocks, 128, 0, stream>>>(nodeB, conv_W + 1 * DGATE * DGATE, A_hat, nodeA);
        gcn_layer_kernel<<<blocks, 128, 0, stream>>>(nodeA, conv_W + 2 * DGATE * DGATE, A_hat, nodeB);
    }

    // K5: router
    router_kernel<<<(T_TOK + 255) / 256, 256, 0, stream>>>(
        nodeB, w_proj, loss_out, wfull, rw_un, sel);

    // K6/K7: per-expert FFN, sequential experts (deterministic accumulate)
    for (int e = 0; e < NEXP; ++e) {
        // h = silu(x @ W1[e])   [2048 x 2048 x 1024]
        wmma_gemm_tiled<<<(T_TOK / BM) * (HID / BN), 256, 0, stream>>>(
            x_bf, DIM, W1 + (size_t)e * DIM * HID, HID, HID, DIM,
            /*mode=*/0, h_bf, nullptr, nullptr, 0, HID);
        // final += w_full[:,e] * (h @ W2[e])   [2048 x 1024 x 2048]
        wmma_gemm_tiled<<<(T_TOK / BM) * (DIM / BN), 256, 0, stream>>>(
            h_bf, HID, W2 + (size_t)e * HID * DIM, DIM, DIM, HID,
            /*mode=*/1, nullptr, final_out, wfull, e, DIM);
    }

    // K8: loss_component tail rows
    finalize_kernel<<<1, 32, 0, stream>>>(count, lamb, theta, topk, rw_un, sel, loss_out);
}
